// LinLayer_bs_24163486007736
// MI455X (gfx1250) — matrix operand order repaired and checked by reading, not compiled, not run on hardware
//
#include <hip/hip_runtime.h>

typedef _Float16 v16h __attribute__((ext_vector_type(16)));
typedef _Float16 h8   __attribute__((ext_vector_type(8)));
typedef float    v8f  __attribute__((ext_vector_type(8)));

#define BATCH 8192
#define INP   1024
#define OUTN  1024
#define NSLICE 4

#define M_TILE 64
#define N_TILE 64
#define K_TILE 32
#define LDP    40   // LDS pitch in halves (32 data + 8 pad), 80B rows, 16B-aligned subtiles

__device__ __forceinline__ float quant128(float x) {
    // quant(x, 8, 1.0): clip to +-(1 - 1/128), round-half-even to 1/128 grid
    float cl = fminf(fmaxf(x, -0.9921875f), 0.9921875f);
    return rintf(cl * 128.0f) * 0.0078125f;
}

// low 32 bits of a generic pointer to LDS == AS(3) byte offset (HW adds LDS_BASE)
__device__ __forceinline__ unsigned lds_addr32(const void* p) {
    return (unsigned)(unsigned long long)p;
}

// global -> LDS async copy, 16B per lane; IOFFSET is added to BOTH addresses
#define ASYNC_B128(ldsa, voff, sbase, IMM)                                   \
    asm volatile("global_load_async_to_lds_b128 %0, %1, %2 offset:" #IMM     \
                 :: "v"(ldsa), "v"(voff), "s"(sbase) : "memory")

// ---------------------------------------------------------------------------
// Prologue 1: activation bit-slicing.  x = clamp(input/a1, -1, 1); 4 residual
// slices; store y_i * 128 (integer in [-127,127]) as fp16, layout [4][B][K].
// ---------------------------------------------------------------------------
__global__ __launch_bounds__(256) void cdna5_prep_act(
    const float* __restrict__ inp, const float* __restrict__ a1p,
    _Float16* __restrict__ As)
{
    const size_t base = ((size_t)blockIdx.x * 256 + threadIdx.x) * 8;
    const float inva = 1.0f / a1p[0];
    const float4 v0 = *(const float4*)(inp + base);
    const float4 v1 = *(const float4*)(inp + base + 4);
    float xs[8] = {v0.x, v0.y, v0.z, v0.w, v1.x, v1.y, v1.z, v1.w};
    const float invb[NSLICE]  = {1.0f, 2.0f, 4.0f, 8.0f};
    const float bstep[NSLICE] = {1.0f/128.0f, 1.0f/256.0f, 1.0f/512.0f, 1.0f/1024.0f};
    h8 q[NSLICE];
    #pragma unroll
    for (int j = 0; j < 8; ++j) {
        float xi = fminf(fmaxf(xs[j] * inva, -1.0f), 1.0f);
        #pragma unroll
        for (int s = 0; s < NSLICE; ++s) {
            float sc = fminf(fmaxf(xi * invb[s], -0.9921875f), 0.9921875f);
            float qq = rintf(sc * 128.0f);           // integer slice value
            q[s][j] = (_Float16)qq;                  // exact in fp16
            xi -= qq * bstep[s];
        }
    }
    #pragma unroll
    for (int s = 0; s < NSLICE; ++s)
        *(h8*)(As + (size_t)s * ((size_t)BATCH * INP) + base) = q[s];
}

// ---------------------------------------------------------------------------
// Prologue 2: weights.  eff_w[s][k][n] = quant8(clamp(w,+-1)) + noise_w[s]
// written TRANSPOSED as fp16 [4][OUT n][INP k] (k contiguous) via LDS tiles.
// ---------------------------------------------------------------------------
__global__ __launch_bounds__(256) void cdna5_prep_w(
    const float* __restrict__ w, const float* __restrict__ nw,
    _Float16* __restrict__ Wt)
{
    __shared__ float    s_wq[64][65];
    __shared__ _Float16 s_h[64][72];    // pitch 72 halves = 144B, 16B-aligned rows

    const int t = threadIdx.x;
    const int kBase = blockIdx.y * 64;
    const int nBase = blockIdx.x * 64;
    const int r  = t >> 2;              // k row within tile
    const int c0 = (t & 3) * 16;        // n start within tile

    #pragma unroll
    for (int j = 0; j < 4; ++j) {
        float4 v = *(const float4*)(w + (size_t)(kBase + r) * OUTN + nBase + c0 + j * 4);
        float tmp[4] = {v.x, v.y, v.z, v.w};
        #pragma unroll
        for (int l = 0; l < 4; ++l)
            s_wq[r][c0 + j * 4 + l] = quant128(tmp[l]);  // clamp(+-1) folded in
    }
    __syncthreads();

    for (int s = 0; s < NSLICE; ++s) {
        #pragma unroll
        for (int j = 0; j < 4; ++j) {
            float4 v = *(const float4*)(nw + (size_t)s * INP * OUTN
                                           + (size_t)(kBase + r) * OUTN + nBase + c0 + j * 4);
            float tmp[4] = {v.x, v.y, v.z, v.w};
            #pragma unroll
            for (int l = 0; l < 4; ++l) {
                int c = c0 + j * 4 + l;
                s_h[c][r] = (_Float16)(s_wq[r][c] + tmp[l]);   // transpose in LDS
            }
        }
        __syncthreads();
        const int n  = t >> 2;
        const int kc = (t & 3) * 16;
        uint4 u0 = *(const uint4*)&s_h[n][kc];
        uint4 u1 = *(const uint4*)&s_h[n][kc + 8];
        uint4* dst = (uint4*)(Wt + (size_t)(s * OUTN + nBase + n) * INP + kBase + kc);
        dst[0] = u0; dst[1] = u1;
        __syncthreads();
    }
}

// ---------------------------------------------------------------------------
// Prologue 3: bias.  eff_b[s][o] = quant8(bias[o]) + noise_b[s][o]  (fp32)
// ---------------------------------------------------------------------------
__global__ __launch_bounds__(256) void cdna5_prep_bias(
    const float* __restrict__ bias, const float* __restrict__ nb,
    float* __restrict__ biasE)
{
    int idx = blockIdx.x * 256 + threadIdx.x;   // 4096 total
    int o = idx & (OUTN - 1);
    biasE[idx] = quant128(bias[o]) + nb[idx];
}

// ---------------------------------------------------------------------------
// Main: fused 4-slice f16 WMMA GEMM + per-slice requant + beta reconstruction.
// Block = 256 threads (8 waves, 4x2), tile 64M x 64N, K-step 32.
// Double-buffered LDS filled by GLOBAL_LOAD_ASYNC_TO_LDS_B128 (no staging
// VGPRs -> no scratch spills); s_wait_asynccnt 8 overlaps tile k+1 loads
// with tile k compute.
// ---------------------------------------------------------------------------
__global__ __launch_bounds__(256) void cdna5_gemm_slices(
    const _Float16* __restrict__ As, const _Float16* __restrict__ Wt,
    const float* __restrict__ biasE, const float* __restrict__ a1p,
    float* __restrict__ out)
{
    __shared__ _Float16 ldsA[2][NSLICE][M_TILE][LDP];   // 2 x 20 KiB
    __shared__ _Float16 ldsW[2][NSLICE][N_TILE][LDP];   // 2 x 20 KiB

    const int t = threadIdx.x;
    const int mBase = blockIdx.y * M_TILE;
    const int nBase = blockIdx.x * N_TILE;

    // async staging: thread owns one (slice, row) 64B run per buffer
    const int ls = t >> 6;
    const int lr = t & 63;
    const unsigned long long baseA = (unsigned long long)As;
    const unsigned long long baseW = (unsigned long long)Wt;
    const unsigned rowOffA = (unsigned)(ls * BATCH + mBase + lr) * (INP * 2);
    const unsigned rowOffW = (unsigned)(ls * OUTN  + nBase + lr) * (INP * 2);
    const unsigned lA[2] = { lds_addr32(&ldsA[0][ls][lr][0]),
                             lds_addr32(&ldsA[1][ls][lr][0]) };
    const unsigned lW[2] = { lds_addr32(&ldsW[0][ls][lr][0]),
                             lds_addr32(&ldsW[1][ls][lr][0]) };

    // wave/lane mapping for WMMA fragments
    const int wid  = t >> 5;
    const int lane = t & 31;
    const int mi   = (wid >> 1) * 16;     // wave's M offset in tile
    const int nj   = (wid & 1) * 32;      // wave's N offset in tile
    const int lm   = lane & 15;
    const int lhi  = lane >> 4;

    const v8f vzero = {0.f, 0.f, 0.f, 0.f, 0.f, 0.f, 0.f, 0.f};
    v8f acc[NSLICE][2];
    #pragma unroll
    for (int s = 0; s < NSLICE; ++s) { acc[s][0] = vzero; acc[s][1] = vzero; }

    const int KSTEPS = INP / K_TILE;      // 32

    // prologue: tile 0 -> buffer 0
    {
        unsigned va = rowOffA, vw = rowOffW;
        ASYNC_B128(lA[0], va, baseA, 0);  ASYNC_B128(lA[0], va, baseA, 16);
        ASYNC_B128(lA[0], va, baseA, 32); ASYNC_B128(lA[0], va, baseA, 48);
        ASYNC_B128(lW[0], vw, baseW, 0);  ASYNC_B128(lW[0], vw, baseW, 16);
        ASYNC_B128(lW[0], vw, baseW, 32); ASYNC_B128(lW[0], vw, baseW, 48);
    }

    for (int kk = 0; kk < KSTEPS; ++kk) {
        const int cur = kk & 1;
        __syncthreads();   // all readers of buffer cur^1 (iter kk-1) are done
        if (kk + 1 < KSTEPS) {
            const int nxt = cur ^ 1;
            unsigned va = rowOffA + (unsigned)(kk + 1) * (K_TILE * 2);
            unsigned vw = rowOffW + (unsigned)(kk + 1) * (K_TILE * 2);
            ASYNC_B128(lA[nxt], va, baseA, 0);  ASYNC_B128(lA[nxt], va, baseA, 16);
            ASYNC_B128(lA[nxt], va, baseA, 32); ASYNC_B128(lA[nxt], va, baseA, 48);
            ASYNC_B128(lW[nxt], vw, baseW, 0);  ASYNC_B128(lW[nxt], vw, baseW, 16);
            ASYNC_B128(lW[nxt], vw, baseW, 32); ASYNC_B128(lW[nxt], vw, baseW, 48);
            // wait for tile kk (8 newer ops for tile kk+1 may stay in flight)
            asm volatile("s_wait_asynccnt 0x8" ::: "memory");
        } else {
            asm volatile("s_wait_asynccnt 0x0" ::: "memory");
        }
        __syncthreads();   // every wave's portion of tile kk is in LDS

        const _Float16 (*bufA)[M_TILE][LDP] = ldsA[cur];
        const _Float16 (*bufW)[N_TILE][LDP] = ldsW[cur];
        #pragma unroll
        for (int s = 0; s < NSLICE; ++s) {
            // A fragment: lane m holds K [lhi*8, +8) and [lhi*8+16, +8)
            h8 alo = *(const h8*)&bufA[s][mi + lm][lhi * 8];
            h8 ahi = *(const h8*)&bufA[s][mi + lm][lhi * 8 + 16];
            v16h af = __builtin_shufflevector(alo, ahi,
                        0,1,2,3,4,5,6,7,8,9,10,11,12,13,14,15);
            #pragma unroll
            for (int nt = 0; nt < 2; ++nt) {
                int col = nj + nt * 16 + lm;
                h8 blo = *(const h8*)&bufW[s][col][lhi * 8];
                h8 bhi = *(const h8*)&bufW[s][col][lhi * 8 + 16];
                v16h bf = __builtin_shufflevector(blo, bhi,
                            0,1,2,3,4,5,6,7,8,9,10,11,12,13,14,15);
                acc[s][nt] = __builtin_amdgcn_wmma_f32_16x16x32_f16(
                    false, af, false, bf, (short)0, acc[s][nt], false, false);
            }
        }
    }

    // epilogue: out = a1 * sum_s beta_s * quant8(acc_s/128 + bias_s)
    const float a1v = a1p[0];
    const float beta[NSLICE] = {1.0f, 0.5f, 0.25f, 0.125f};
    #pragma unroll
    for (int nt = 0; nt < 2; ++nt) {
        const int n = nBase + nj + nt * 16 + lm;
        float bl[NSLICE];
        #pragma unroll
        for (int s = 0; s < NSLICE; ++s) bl[s] = biasE[s * OUTN + n];
        #pragma unroll
        for (int e = 0; e < 8; ++e) {
            const int row = mBase + mi + e + lhi * 8;
            float r = 0.0f;
            #pragma unroll
            for (int s = 0; s < NSLICE; ++s) {
                float o = acc[s][nt][e] * 0.0078125f + bl[s];
                r += beta[s] * quant128(o);
            }
            out[(size_t)row * OUTN + n] = r * a1v;
        }
    }
}

// ---------------------------------------------------------------------------
extern "C" void kernel_launch(void* const* d_in, const int* in_sizes, int n_in,
                              void* d_out, int out_size, void* d_ws, size_t ws_size,
                              hipStream_t stream) {
    const float* inp  = (const float*)d_in[0];   // [8192,1024]
    const float* w    = (const float*)d_in[1];   // [1,1024,1024]
    const float* bias = (const float*)d_in[2];   // [1,1,1024]
    const float* a1   = (const float*)d_in[3];   // [1]
    const float* nw   = (const float*)d_in[4];   // [4,1024,1024]
    const float* nb   = (const float*)d_in[5];   // [4,1,1024]
    float* out = (float*)d_out;

    char* ws = (char*)d_ws;
    _Float16* As = (_Float16*)ws;                                          // 64 MiB
    _Float16* Wt = (_Float16*)(ws + (size_t)NSLICE * BATCH * INP * 2);     // 8 MiB
    float* biasE = (float*)(ws + (size_t)NSLICE * BATCH * INP * 2
                               + (size_t)NSLICE * INP * OUTN * 2);         // 16 KiB

    cdna5_prep_act<<<(BATCH * INP) / (256 * 8), 256, 0, stream>>>(inp, a1, As);
    cdna5_prep_w<<<dim3(OUTN / 64, INP / 64), 256, 0, stream>>>(w, nw, Wt);
    cdna5_prep_bias<<<(NSLICE * OUTN) / 256, 256, 0, stream>>>(bias, nb, biasE);
    cdna5_gemm_slices<<<dim3(OUTN / N_TILE, BATCH / M_TILE), 256, 0, stream>>>(
        As, Wt, biasE, a1, out);
}
